// RecognitionLattice_3547642987130
// MI455X (gfx1250) — matrix operand order repaired and checked by reading, not compiled, not run on hardware
//
#include <hip/hip_runtime.h>
#include <hip/hip_bf16.h>
#include <math.h>

// Problem constants (fixed by the reference setup_inputs()).
#define B_  8
#define T_  512
#define D_  256
#define H_  256
#define V_  32
#define U_  64
#define C_  33          // V+1 context states
#define BT_ (B_*T_)     // 4096
#define ARCW 36         // padded row stride for arc[bt][c][.]: cols 0..31 = lex, 32 = blank
#define NEGV (-1e30f)

typedef __attribute__((ext_vector_type(16))) _Float16 v16h;
typedef __attribute__((ext_vector_type(8)))  float    v8f;
typedef __attribute__((ext_vector_type(2)))  _Float16 h2;

// ---- per-lane K offset for 16-bit A fragment (16x32), ISA 7.12.2 ----
// lane group g = lane/16; VGPR j holds K = base(j)+8g, base(j)+8g+1
__device__ __forceinline__ int ka_off(int j, int g) {
    return ((j < 4) ? (2 * j) : (16 + 2 * (j - 4))) + 8 * g;
}
// ---- 16-bit B fragment (32x16) as loaded in this file (A uses the same order): element i of v16h holds K = i + 16g ----

__device__ __forceinline__ float logaddexp_f(float a, float b) {
    float m = fmaxf(a, b);
    float d = fminf(a, b) - m;                 // <= 0
    return m + __logf(1.f + __expf(d));        // both==NEGV -> ~NEGV, one==NEGV -> m
}

// ============================================================================
// Stage 0: pack weights to f16.
//   wfT[h][d]   = (f16) w_frame[d][h]              (256x256, transposed)
//   waugT[n][h] = (f16) [w_lex[h][n] (n<32) | v_blank[h] (n==32) | 0] (48x256)
// ============================================================================
__global__ __launch_bounds__(256) void k_prep(const float* __restrict__ w_frame,
                                              const float* __restrict__ v_blank,
                                              const float* __restrict__ w_lex,
                                              _Float16* __restrict__ wfT,
                                              _Float16* __restrict__ waugT) {
    int idx = blockIdx.x * 256 + threadIdx.x;           // 0 .. 65535
    if (idx < H_ * D_) {
        int h = idx >> 8, d = idx & 255;                // wfT[h*256 + d]
        wfT[idx] = (_Float16)w_frame[d * H_ + h];
    }
    if (idx < 48 * H_) {
        int n = idx >> 8, h = idx & 255;
        float v = (n < V_) ? w_lex[h * V_ + n] : ((n == V_) ? v_blank[h] : 0.f);
        waugT[idx] = (_Float16)v;
    }
}

// ============================================================================
// Stage 1: fp[bt][h] = frames[bt][:] @ w_frame[:][h]   (4096x256x256 GEMM)
// One wave per 16x16 output tile; K loop of v_wmma_f32_16x16x32_f16.
// ============================================================================
__global__ __launch_bounds__(32) void k_fp_gemm(const float* __restrict__ frames,
                                                const _Float16* __restrict__ wfT,
                                                float* __restrict__ fp) {
    const int mt = blockIdx.x;            // 0..255  (BT/16)
    const int nt = blockIdx.y;            // 0..15   (H/16)
    const int lane = threadIdx.x;
    const int g = lane >> 4, l16 = lane & 15;
    const int arow = mt * 16 + l16;       // A row this lane supplies
    const int bcol = nt * 16 + l16;       // B column this lane supplies

    const float*    ap = frames + (size_t)arow * D_;
    const _Float16* bp = wfT    + (size_t)bcol * D_;

    v8f acc = {};
    for (int kb = 0; kb < D_; kb += 32) {
        v16h a, b;
#pragma unroll
        for (int j = 0; j < 8; ++j) {
            int ka = kb + 2 * j + 16 * g;
            float2 av = *(const float2*)(ap + ka);          // 8B aligned (ka even)
            a[2 * j]     = (_Float16)av.x;
            a[2 * j + 1] = (_Float16)av.y;
            int kbi = kb + 2 * j + 16 * g;                  // B: K = i + 16g
            h2 bv = *(const h2*)(bp + kbi);                 // 4B aligned
            b[2 * j]     = bv.x;
            b[2 * j + 1] = bv.y;
        }
        acc = __builtin_amdgcn_wmma_f32_16x16x32_f16(false, a, false, b,
                                                     (short)0, acc, false, false);
    }
    // D layout: VGPR r -> M = r + 8g, N = l16
#pragma unroll
    for (int r = 0; r < 8; ++r) {
        int m = mt * 16 + r + 8 * g;
        fp[(size_t)m * H_ + nt * 16 + l16] = acc[r];
    }
}

// ============================================================================
// Stage 2: arc[bt][c][v<32] = lex, arc[bt][c][32] = blank
//   hidden[r][h] = tanh(fp[bt][h] + ctx_emb[c][h])  (f16, in LDS)
//   arc row = hidden @ [w_lex | v_blank]  via WMMA, N padded to 48 (3 tiles)
// One wave per (16-row bt tile, context state c).
// ============================================================================
__global__ __launch_bounds__(32) void k_arc_gemm(const float* __restrict__ fp,
                                                 const float* __restrict__ ctx_emb,
                                                 const _Float16* __restrict__ waugT,
                                                 float* __restrict__ arc) {
    __shared__ _Float16 hid[16][H_];      // 8 KB
    const int mt = blockIdx.x;            // 0..255
    const int c  = blockIdx.y;            // 0..32
    const int lane = threadIdx.x;

    // Build the tanh-hidden tile (16 x 256) in f16.
    for (int idx = lane; idx < 16 * H_; idx += 32) {
        int r = idx >> 8, h = idx & 255;
        float v = fp[(size_t)(mt * 16 + r) * H_ + h] + ctx_emb[c * H_ + h];
        hid[r][h] = (_Float16)tanhf(v);
    }
    __syncthreads();

    const int g = lane >> 4, l16 = lane & 15;
    v8f acc0 = {}, acc1 = {}, acc2 = {};
    for (int kb = 0; kb < H_; kb += 32) {
        v16h a;
#pragma unroll
        for (int j = 0; j < 8; ++j) {
            int ka = kb + 2 * j + 16 * g;
            h2 av = *(const h2*)&hid[l16][ka];
            a[2 * j]     = av.x;
            a[2 * j + 1] = av.y;
        }
#pragma unroll
        for (int n = 0; n < 3; ++n) {
            v16h b;
            const _Float16* bp = waugT + (size_t)(n * 16 + l16) * H_ + kb;
#pragma unroll
            for (int j = 0; j < 8; ++j) {
                h2 bv = *(const h2*)(bp + 2 * j + 16 * g);
                b[2 * j]     = bv.x;
                b[2 * j + 1] = bv.y;
            }
            v8f* accp = (n == 0) ? &acc0 : (n == 1) ? &acc1 : &acc2;
            *accp = __builtin_amdgcn_wmma_f32_16x16x32_f16(false, a, false, b,
                                                           (short)0, *accp, false, false);
        }
    }
    // Store only the 33 valid columns (32 lex + 1 blank).
#pragma unroll
    for (int n = 0; n < 3; ++n) {
        int col = n * 16 + l16;
        if (col < C_) {
            const v8f& acc = (n == 0) ? acc0 : (n == 1) ? acc1 : acc2;
#pragma unroll
            for (int r = 0; r < 8; ++r) {
                int bt = mt * 16 + r + 8 * g;
                arc[((size_t)bt * C_ + c) * ARCW + col] = acc[r];
            }
        }
    }
}

// ============================================================================
// Stage 3: log-semiring scans (denominator over C states, numerator over U+1
// prefix states). One block per batch element; α/β ping-pong in LDS.
// ============================================================================
__global__ __launch_bounds__(160) void k_scan(const float* __restrict__ arc,
                                              const int* __restrict__ num_frames,
                                              const int* __restrict__ labels,
                                              const int* __restrict__ num_labels,
                                              float* __restrict__ out) {
    __shared__ float alpha[2][C_];
    __shared__ float beta[2][U_ + 1];
    __shared__ int   cs[U_ + 1];          // context state per prefix position
    __shared__ int   li[U_];              // lex column (label-1, clipped)

    const int b   = blockIdx.x;
    const int tid = threadIdx.x;
    const int nf  = num_frames[b];
    const int nl  = num_labels[b];

    if (tid < U_ + 1) {
        cs[tid] = (tid == 0) ? 0 : labels[b * U_ + tid - 1];
        beta[0][tid] = (tid == 0) ? 0.f : NEGV;
    }
    if (tid < C_) alpha[0][tid] = (tid == 0) ? 0.f : NEGV;
    if (tid < U_) {
        int lv = labels[b * U_ + tid] - 1;
        li[tid] = (lv < 0) ? 0 : ((lv > V_ - 1) ? V_ - 1 : lv);
    }
    __syncthreads();

    int cur = 0;
    for (int t = 0; t < nf; ++t) {        // frame mask is a prefix: just stop at nf
        const float* arcT = arc + ((size_t)(b * T_ + t) * C_) * ARCW;
        const int nxt = cur ^ 1;

        if (tid < C_) {
            // --------- denominator over context states ---------
            float stay = alpha[cur][tid] + arcT[tid * ARCW + V_];   // blank
            float nv;
            if (tid == 0) {
                nv = stay;
            } else {
                int v = tid - 1;                                    // dest state v+1
                float m = NEGV, s = 0.f;                            // streaming LSE over c
                for (int c = 0; c < C_; ++c) {
                    float x = alpha[cur][c] + arcT[c * ARCW + v];
                    if (x > m) { s = s * __expf(m - x) + 1.f; m = x; }
                    else       { s += __expf(x - m); }
                }
                nv = logaddexp_f(stay, m + __logf(s));
            }
            alpha[nxt][tid] = nv;
        } else if (tid >= 64 && tid < 64 + U_ + 1) {
            // --------- numerator over label-prefix states ---------
            int u = tid - 64;
            float stay = beta[cur][u] + arcT[cs[u] * ARCW + V_];    // blank at ctx state
            float nv;
            if (u == 0) {
                nv = stay;
            } else {
                int up = u - 1;
                float lexn = (up < nl) ? arcT[cs[up] * ARCW + li[up]] : NEGV;
                nv = logaddexp_f(stay, beta[cur][up] + lexn);
            }
            beta[nxt][u] = nv;
        }
        __syncthreads();
        cur = nxt;
    }

    if (tid == 0) {
        float m = NEGV, s = 0.f;
        for (int c = 0; c < C_; ++c) {
            float x = alpha[cur][c];
            if (x > m) { s = s * __expf(m - x) + 1.f; m = x; }
            else       { s += __expf(x - m); }
        }
        float den = m + __logf(s);
        float num = beta[cur][nl];
        out[b] = den - num;
    }
}

// ============================================================================
extern "C" void kernel_launch(void* const* d_in, const int* in_sizes, int n_in,
                              void* d_out, int out_size, void* d_ws, size_t ws_size,
                              hipStream_t stream) {
    const float* frames     = (const float*)d_in[0];
    const int*   num_frames = (const int*)  d_in[1];
    const int*   labels     = (const int*)  d_in[2];
    const int*   num_labels = (const int*)  d_in[3];
    const float* w_frame    = (const float*)d_in[4];
    const float* ctx_emb    = (const float*)d_in[5];
    const float* v_blank    = (const float*)d_in[6];
    const float* w_lex      = (const float*)d_in[7];

    // Workspace layout (all L2-resident: ~24 MB total vs 192 MB L2)
    char* ws = (char*)d_ws;
    float*    fp    = (float*)ws;                                   // 4096*256*4   = 4 MB
    _Float16* wfT   = (_Float16*)(ws + (size_t)BT_ * H_ * 4);       // 65536*2      = 128 KB
    _Float16* waugT = (_Float16*)((char*)wfT + (size_t)H_ * D_ * 2);// 48*256*2     = 24 KB
    float*    arcb  = (float*)((char*)waugT + 48 * H_ * 2);         // 4096*33*36*4 ≈ 19 MB

    k_prep    <<<dim3(256),       256, 0, stream>>>(w_frame, v_blank, w_lex, wfT, waugT);
    k_fp_gemm <<<dim3(BT_/16, H_/16), 32, 0, stream>>>(frames, wfT, fp);
    k_arc_gemm<<<dim3(BT_/16, C_),    32, 0, stream>>>(fp, ctx_emb, waugT, arcb);
    k_scan    <<<dim3(B_),       160, 0, stream>>>(arcb, num_frames, labels, num_labels,
                                                   (float*)d_out);
}
